// MixtureOfExpertsLayer_57681410785521
// MI455X (gfx1250) — matrix operand order repaired and checked by reading, not compiled, not run on hardware
//
#include <hip/hip_runtime.h>
#include <cstdint>
#include <cstddef>

// ---------------------------------------------------------------------------
// MoE layer for MI455X (gfx1250), wave32 + WMMA bf16 + async global->LDS DMA.
//   T=16384 tokens, H=1024, 2H=2048, E=8, top-2 routing.
// Pipeline: zero-acc -> router(top2) -> deterministic per-expert compaction ->
//           fused expert FFN (async x-gather -> WMMA GEMM1 -> GELU -> LDS ->
//           WMMA GEMM2 -> weighted atomic scatter) -> residual+LayerNorm.
// ---------------------------------------------------------------------------

typedef __attribute__((ext_vector_type(16))) __bf16 v16bf;
typedef __attribute__((ext_vector_type(8)))  __bf16 v8bf;
typedef __attribute__((ext_vector_type(8)))  float  v8f;

#define T_TOK 16384
#define HDIM  1024
#define H2    2048
#define NEXP  8
#define MT    32                       // tokens per expert tile
#define TILES_PER_E (T_TOK / MT)       // 512

// ---- gfx1250 async global->LDS DMA (ASYNCcnt-tracked, ISA §10/§15.18) -----
// Inline asm per the VGLOBAL encoding: VDST = per-lane LDS byte offset,
// VADDR = 64-bit global address. Generic->LDS offset: flat aperture rule
// the LDS offset is the low 32 bits of the address.
__device__ __forceinline__ void async_cp16(const void* gsrc, void* ldst) {
  unsigned lds_off = (unsigned)(uintptr_t)ldst;
  unsigned long long gaddr = (unsigned long long)(uintptr_t)gsrc;
  asm volatile("global_load_async_to_lds_b128 %0, %1, off"
               :
               : "v"(lds_off), "v"(gaddr)
               : "memory");
}
__device__ __forceinline__ void wait_async0() {
  asm volatile("s_wait_asynccnt 0" ::: "memory");
}

// float -> bf16 (RNE). Native fptrunc lets the backend use v_cvt_pk_bf16_f32.
__device__ __forceinline__ __bf16 f2bf(float f) { return (__bf16)f; }

__device__ __forceinline__ float gelu_exact(float x) {
  return 0.5f * x * (1.0f + erff(x * 0.70710678118654752f));
}

__device__ __forceinline__ v8f wmma_bf16(v16bf a, v16bf b, v8f c) {
  // emits v_wmma_f32_16x16x32_bf16
  return __builtin_amdgcn_wmma_f32_16x16x32_bf16(
      /*neg_a=*/false, a, /*neg_b=*/false, b,
      /*c_mod=*/(short)0, c, /*reuse_a=*/false, /*reuse_b=*/false);
}

// A fragment: 16x32 bf16 tile from LDS, row-major [.. x ld].
// ISA 7.12.2: lane<16 row M=lane holds K 0..7 & 16..23; lane>=16 row M=lane-16
// holds K 8..15 & 24..31. Two 16B ds loads per lane.
__device__ __forceinline__ v16bf load_A(const __bf16* base, int ld, int rowOff,
                                        int k0, int lane) {
  int r  = (lane & 15) + rowOff;
  int hi = (lane >> 4) & 1;
  const __bf16* row = base + (size_t)r * ld + k0;
  v8bf lo = *(const v8bf*)(row + hi * 8);
  v8bf hv = *(const v8bf*)(row + 16 + hi * 8);
  v16bf a;
#pragma unroll
  for (int p = 0; p < 8; ++p) { a[p] = lo[p]; a[p + 8] = hv[p]; }
  return a;
}

// B fragment: 32x16 tile of a row-major fp32 weight [K x ldw], converted to
// bf16 in-register. Lane l supplies column N=n0+l%16; lane<16 holds K 0..7 &
// 16..23, lane>=16 holds K 8..15 & 24..31.
__device__ __forceinline__ v16bf load_B_g(const float* __restrict__ W, int ldw,
                                          int k0, int n0, int lane) {
  const float* row = W + (size_t)k0 * ldw + n0 + (lane & 15);
  int hi = (lane >> 4) & 1;
  __builtin_prefetch(row + (size_t)32 * ldw, 0, 0);  // next K-block (speculative)
  v16bf b;
#pragma unroll
  for (int p = 0; p < 8; ++p) {
    b[p]     = f2bf(row[(size_t)(hi * 8 + p) * ldw]);
    b[p + 8] = f2bf(row[(size_t)(16 + hi * 8 + p) * ldw]);
  }
  return b;
}

// ---------------------------------------------------------------------------
// K0: zero the fp32 output accumulator (atomically accumulated later).
// ---------------------------------------------------------------------------
__global__ void k_zero(float* __restrict__ p, int n4) {
  int i = blockIdx.x * blockDim.x + threadIdx.x;
  if (i < n4) ((float4*)p)[i] = make_float4(0.f, 0.f, 0.f, 0.f);
}

// ---------------------------------------------------------------------------
// K1: router. One wave (32 lanes) per token; Wr (8x1024 fp32, 32KB) in LDS.
// ---------------------------------------------------------------------------
__global__ void k_router(const float* __restrict__ X, const float* __restrict__ Wr,
                         int* __restrict__ tki, float* __restrict__ tkw) {
  __shared__ float sWr[NEXP * HDIM];
  int tid = threadIdx.x;
  for (int i = tid * 4; i < NEXP * HDIM; i += blockDim.x * 4)
    *(float4*)(sWr + i) = *(const float4*)(Wr + i);
  __syncthreads();

  int lane = tid & 31, wid = tid >> 5;
  int t = blockIdx.x * 8 + wid;
  const float* x = X + (size_t)t * HDIM;

  float acc[NEXP];
#pragma unroll
  for (int e = 0; e < NEXP; ++e) acc[e] = 0.f;

  for (int i = 0; i < HDIM; i += 128) {
    float4 xv = *(const float4*)(x + i + lane * 4);
#pragma unroll
    for (int e = 0; e < NEXP; ++e) {
      float4 wv = *(const float4*)(sWr + e * HDIM + i + lane * 4);
      acc[e] += xv.x * wv.x + xv.y * wv.y + xv.z * wv.z + xv.w * wv.w;
    }
  }
#pragma unroll
  for (int e = 0; e < NEXP; ++e)
#pragma unroll
    for (int off = 16; off > 0; off >>= 1)
      acc[e] += __shfl_xor(acc[e], off, 32);

  if (lane == 0) {
    int i0 = 0; float v0 = acc[0];
    for (int e = 1; e < NEXP; ++e) if (acc[e] > v0) { v0 = acc[e]; i0 = e; }
    int i1 = (i0 == 0) ? 1 : 0; float v1 = acc[i1];
    for (int e = 0; e < NEXP; ++e)
      if (e != i0 && acc[e] > v1) { v1 = acc[e]; i1 = e; }
    // softmax over the two selected logits (v0 >= v1)
    float e0 = 1.0f, e1 = __expf(v1 - v0);
    float s = e0 + e1;
    tki[2 * t] = i0; tki[2 * t + 1] = i1;
    tkw[2 * t] = e0 / s; tkw[2 * t + 1] = e1 / s;
  }
}

// ---------------------------------------------------------------------------
// K2: deterministic compaction. One block per expert; ballot/popc wave32 scan
// builds ordered token lists (no atomics -> bit-stable every launch).
// ---------------------------------------------------------------------------
__global__ void k_compact(const int* __restrict__ tki, const float* __restrict__ tkw,
                          int* __restrict__ tok_list, float* __restrict__ w_list,
                          int* __restrict__ counts) {
  int e = blockIdx.x;
  int tid = threadIdx.x, lane = tid & 31, wid = tid >> 5;
  __shared__ int wsum[8];
  int total = 0;
  for (int base = 0; base < T_TOK; base += 256) {
    int t = base + tid;
    int flag = 0; float wt = 0.f;
    int i0 = tki[2 * t], i1 = tki[2 * t + 1];
    if (i0 == e)      { flag = 1; wt = tkw[2 * t]; }
    else if (i1 == e) { flag = 1; wt = tkw[2 * t + 1]; }
    unsigned bal = (unsigned)__ballot(flag);
    int prefix = __popc(bal & ((1u << lane) - 1u));
    if (lane == 0) wsum[wid] = __popc(bal);
    __syncthreads();
    int wbase = 0;
    for (int i = 0; i < wid; ++i) wbase += wsum[i];
    int bsum = 0;
    for (int i = 0; i < 8; ++i) bsum += wsum[i];
    if (flag) {
      int pos = total + wbase + prefix;
      tok_list[e * T_TOK + pos] = t;
      w_list[e * T_TOK + pos] = wt;
    }
    total += bsum;
    __syncthreads();
  }
  if (tid == 0) counts[e] = total;
}

// ---------------------------------------------------------------------------
// K3: fused expert FFN per (expert, 32-token tile). 256 threads = 8 waves.
// LDS: x tile 32x1024 bf16 (64KB) + h tile 32x2048 bf16 (128KB) = 192KB,
// exploiting the 320KB WGP LDS so h never touches HBM. The fp32 staging for
// the async x-gather aliases the (not-yet-written) h region.
// ---------------------------------------------------------------------------
__global__ void k_expert_ffn(const float* __restrict__ X,
                             const float* __restrict__ W1, const float* __restrict__ b1,
                             const float* __restrict__ W2, const float* __restrict__ b2,
                             const int* __restrict__ tok_list,
                             const float* __restrict__ w_list,
                             const int* __restrict__ counts,
                             float* __restrict__ out_acc) {
  extern __shared__ __align__(16) char smem[];
  __bf16* x_lds = (__bf16*)smem;                           // [32][1024]
  __bf16* h_lds = (__bf16*)(smem + (size_t)MT * HDIM * 2); // [32][2048]
  __shared__ int   sTok[MT];
  __shared__ float sWt[MT];

  int e    = blockIdx.x / TILES_PER_E;
  int tile = blockIdx.x % TILES_PER_E;
  int cnt  = counts[e];
  int start = tile * MT;
  if (start >= cnt) return;  // uniform across block

  int tid = threadIdx.x, lane = tid & 31, wid = tid >> 5;

  if (tid < MT) {
    int r = start + tid;
    if (r < cnt) { sTok[tid] = tok_list[e * T_TOK + r]; sWt[tid] = w_list[e * T_TOK + r]; }
    else         { sTok[tid] = 0;                        sWt[tid] = 0.f; }  // weight-0 pad
  }
  __syncthreads();

  // ---- Gather 32 token rows of x into bf16 LDS tile via async DMA ---------
  {
    float* stage = (float*)h_lds;  // 64KB fp32 staging, aliased with h region
    for (int s = 0; s < MT; s += 16) {
#pragma unroll
      for (int i = 0; i < 16; ++i) {
        // each issue copies one full 4KB row: 256 lanes x 16B
        const float* src = X + (size_t)sTok[s + i] * HDIM + tid * 4;
        async_cp16(src, stage + i * HDIM + tid * 4);
      }
      wait_async0();
      __syncthreads();
#pragma unroll
      for (int i = 0; i < 16; ++i) {
        float4 f = *(const float4*)(stage + i * HDIM + tid * 4);
        __bf16* dst = x_lds + (size_t)(s + i) * HDIM + tid * 4;
        dst[0] = f2bf(f.x); dst[1] = f2bf(f.y);
        dst[2] = f2bf(f.z); dst[3] = f2bf(f.w);
      }
      __syncthreads();
    }
  }

  const float* W1e = W1 + (size_t)e * HDIM * H2;
  const float* b1e = b1 + (size_t)e * H2;
  const float* W2e = W2 + (size_t)e * H2 * HDIM;
  const float* b2e = b2 + (size_t)e * HDIM;

  // ---- GEMM1 (32 x 2048 = x @ W1) + bias + exact GELU -> h_lds (bf16) ----
  for (int it = 0; it < 16; ++it) {
    int n1 = it * 128 + wid * 16;  // this wave's 16-column slab
    v8f c0 = {}, c1 = {};
    for (int k = 0; k < HDIM; k += 32) {
      v16bf a0 = load_A(x_lds, HDIM, 0,  k, lane);
      v16bf a1 = load_A(x_lds, HDIM, 16, k, lane);
      v16bf bB = load_B_g(W1e, H2, k, n1, lane);
      c0 = wmma_bf16(a0, bB, c0);
      c1 = wmma_bf16(a1, bB, c1);
    }
    int col = n1 + (lane & 15);
    float bias = b1e[col];
#pragma unroll
    for (int i = 0; i < 8; ++i) {
      int r0 = (lane < 16) ? i : i + 8;       // C layout: VGPR i -> M=i / M=8+i
      h_lds[(size_t)r0 * H2 + col]        = f2bf(gelu_exact(c0[i] + bias));
      h_lds[(size_t)(r0 + 16) * H2 + col] = f2bf(gelu_exact(c1[i] + bias));
    }
  }
  __syncthreads();

  // ---- GEMM2 (32 x 1024 = h @ W2), wave owns a 128-column slab ----
  v8f acc[2][8];
#pragma unroll
  for (int mh = 0; mh < 2; ++mh)
#pragma unroll
    for (int j = 0; j < 8; ++j) acc[mh][j] = (v8f){};

  int n2base = wid * 128;
  for (int k = 0; k < H2; k += 32) {
    v16bf a0 = load_A(h_lds, H2, 0,  k, lane);
    v16bf a1 = load_A(h_lds, H2, 16, k, lane);
#pragma unroll
    for (int j = 0; j < 8; ++j) {
      v16bf bB = load_B_g(W2e, HDIM, k, n2base + j * 16, lane);
      acc[0][j] = wmma_bf16(a0, bB, acc[0][j]);
      acc[1][j] = wmma_bf16(a1, bB, acc[1][j]);
    }
  }

  // ---- Epilogue: +b2, scale by router weight, scatter-add (fp32 atomics) ----
#pragma unroll
  for (int j = 0; j < 8; ++j) {
    int col = n2base + j * 16 + (lane & 15);
    float bias = b2e[col];
#pragma unroll
    for (int mh = 0; mh < 2; ++mh) {
#pragma unroll
      for (int i = 0; i < 8; ++i) {
        int row = ((lane < 16) ? i : i + 8) + mh * 16;
        float val = (acc[mh][j][i] + bias) * sWt[row];
        atomicAdd(out_acc + (size_t)sTok[row] * HDIM + col, val);
      }
    }
  }
}

// ---------------------------------------------------------------------------
// K4: residual + LayerNorm. One block per token row; one float4 per thread.
// ---------------------------------------------------------------------------
__global__ void k_ln(const float* __restrict__ X, const float* __restrict__ acc,
                     const float* __restrict__ gamma, const float* __restrict__ beta,
                     float* __restrict__ out) {
  int t = blockIdx.x;
  int tid = threadIdx.x, lane = tid & 31, wid = tid >> 5;
  __shared__ float s1[8], s2[8];

  const float* x = X + (size_t)t * HDIM;
  const float* a = acc + (size_t)t * HDIM;
  float4 xv = *(const float4*)(x + tid * 4);
  float4 av = *(const float4*)(a + tid * 4);
  float4 r;
  r.x = xv.x + av.x; r.y = xv.y + av.y; r.z = xv.z + av.z; r.w = xv.w + av.w;

  float s = r.x + r.y + r.z + r.w;
  float q = r.x * r.x + r.y * r.y + r.z * r.z + r.w * r.w;
#pragma unroll
  for (int off = 16; off > 0; off >>= 1) {
    s += __shfl_xor(s, off, 32);
    q += __shfl_xor(q, off, 32);
  }
  if (lane == 0) { s1[wid] = s; s2[wid] = q; }
  __syncthreads();
  float sum = 0.f, sq = 0.f;
#pragma unroll
  for (int i = 0; i < 8; ++i) { sum += s1[i]; sq += s2[i]; }

  float mu  = sum * (1.0f / HDIM);
  float var = sq * (1.0f / HDIM) - mu * mu;
  float inv = rsqrtf(var + 1e-5f);

  float4 g  = *(const float4*)(gamma + tid * 4);
  float4 bb = *(const float4*)(beta + tid * 4);
  float4 o;
  o.x = (r.x - mu) * inv * g.x + bb.x;
  o.y = (r.y - mu) * inv * g.y + bb.y;
  o.z = (r.z - mu) * inv * g.z + bb.z;
  o.w = (r.w - mu) * inv * g.w + bb.w;
  *(float4*)(out + (size_t)t * HDIM + tid * 4) = o;
}

// ---------------------------------------------------------------------------
extern "C" void kernel_launch(void* const* d_in, const int* in_sizes, int n_in,
                              void* d_out, int out_size, void* d_ws, size_t ws_size,
                              hipStream_t stream) {
  (void)in_sizes; (void)n_in; (void)out_size; (void)ws_size;

  const float* X     = (const float*)d_in[0];
  const float* Wr    = (const float*)d_in[1];
  const float* W1    = (const float*)d_in[2];
  const float* b1    = (const float*)d_in[3];
  const float* W2    = (const float*)d_in[4];
  const float* b2    = (const float*)d_in[5];
  const float* gamma = (const float*)d_in[6];
  const float* beta  = (const float*)d_in[7];
  float* out = (float*)d_out;

  char* ws = (char*)d_ws;
  float* out_acc  = (float*)ws; ws += (size_t)T_TOK * HDIM * sizeof(float);   // 64 MB
  int*   tki      = (int*)ws;   ws += (size_t)T_TOK * 2 * sizeof(int);
  float* tkw      = (float*)ws; ws += (size_t)T_TOK * 2 * sizeof(float);
  int*   tok_list = (int*)ws;   ws += (size_t)NEXP * T_TOK * sizeof(int);
  float* w_list   = (float*)ws; ws += (size_t)NEXP * T_TOK * sizeof(float);
  int*   counts   = (int*)ws;   ws += 64;

  int n4 = T_TOK * HDIM / 4;
  k_zero<<<(n4 + 255) / 256, 256, 0, stream>>>(out_acc, n4);
  k_router<<<T_TOK / 8, 256, 0, stream>>>(X, Wr, tki, tkw);
  k_compact<<<NEXP, 256, 0, stream>>>(tki, tkw, tok_list, w_list, counts);

  size_t smem = (size_t)MT * HDIM * 2 + (size_t)MT * H2 * 2;  // 192 KB dynamic LDS
  k_expert_ffn<<<NEXP * TILES_PER_E, 256, smem, stream>>>(
      X, W1, b1, W2, b2, tok_list, w_list, counts, out_acc);

  k_ln<<<T_TOK, 256, 0, stream>>>(X, out_acc, gamma, beta, out);
}
